// MultiheadSelfAttention_52304111731338
// MI455X (gfx1250) — matrix operand order repaired and checked by reading, not compiled, not run on hardware
//
#include <hip/hip_runtime.h>
#include <hip/hip_bf16.h>

typedef __attribute__((ext_vector_type(16))) _Float16 v16h;
typedef __attribute__((ext_vector_type(8)))  _Float16 v8h;
typedef __attribute__((ext_vector_type(8)))  float    v8f;

#define D_MODEL   1024
#define NUM_HEADS 16
#define DK        64
#define BATCH     2
#define SEQ       2048
#define MTOT      (BATCH * SEQ)   // 4096

// ---------------------------------------------------------------------------
// GEMM: C[M,N] = A[M,K] * W[N,K]^T   (both A and W are K-contiguous row-major)
// WG = 256 threads = 8 waves (2x4). Wave tile 32x32 (2x2 WMMA), WG tile 64x128.
// fp32 -> f16 conversion happens during LDS staging; f32 accumulate in WMMA.
// ---------------------------------------------------------------------------
__global__ __launch_bounds__(256)
void wmma_gemm_nt(const float* __restrict__ A, const float* __restrict__ W,
                  float* __restrict__ C, int M, int N, int K) {
  __shared__ _Float16 sA[64 * 32];    // [m][k]
  __shared__ _Float16 sB[128 * 32];   // [n][k]
  const int tid  = threadIdx.x;
  const int lane = tid & 31;
  const int w    = tid >> 5;
  const int wm   = w >> 2;            // 0..1
  const int wn   = w & 3;             // 0..3
  const int m0   = blockIdx.y * 64;
  const int n0   = blockIdx.x * 128;
  const int row  = lane & 15;
  const int hif  = lane >> 4;

  v8f acc[2][2] = {};

  for (int kk = 0; kk < K; kk += 32) {
    __syncthreads();
    // stage A tile: 64x32 halfs, 8 per thread
    {
      int o = tid * 8; int m = o >> 5; int k = o & 31;
      const float4* src = (const float4*)(A + (size_t)(m0 + m) * K + kk + k);
      float4 f0 = src[0], f1 = src[1];
      v8h hv;
      hv[0] = (_Float16)f0.x; hv[1] = (_Float16)f0.y;
      hv[2] = (_Float16)f0.z; hv[3] = (_Float16)f0.w;
      hv[4] = (_Float16)f1.x; hv[5] = (_Float16)f1.y;
      hv[6] = (_Float16)f1.z; hv[7] = (_Float16)f1.w;
      *(v8h*)(sA + o) = hv;
    }
    // stage B tile: 128x32 halfs, 16 per thread
    for (int c = 0; c < 2; ++c) {
      int o = (tid + c * 256) * 8; int n = o >> 5; int k = o & 31;
      const float4* src = (const float4*)(W + (size_t)(n0 + n) * K + kk + k);
      float4 f0 = src[0], f1 = src[1];
      v8h hv;
      hv[0] = (_Float16)f0.x; hv[1] = (_Float16)f0.y;
      hv[2] = (_Float16)f0.z; hv[3] = (_Float16)f0.w;
      hv[4] = (_Float16)f1.x; hv[5] = (_Float16)f1.y;
      hv[6] = (_Float16)f1.z; hv[7] = (_Float16)f1.w;
      *(v8h*)(sB + o) = hv;
    }
    if (kk + 32 < K) {  // next-tile prefetch -> global_prefetch_b8
      __builtin_prefetch(A + (size_t)(m0 + (tid >> 2)) * K + kk + 32 + ((tid & 3) << 3), 0, 1);
      __builtin_prefetch(W + (size_t)(n0 + (tid >> 1)) * K + kk + 32 + ((tid & 1) << 4), 0, 1);
    }
    __syncthreads();

    // A fragments: lanes 0-15 hold K 0..15, lanes 16-31 hold K 16..31 (contiguous)
    v16h afr[2], bfr[2];
    for (int mb = 0; mb < 2; ++mb) {
      const _Float16* p = sA + (wm * 32 + mb * 16 + row) * 32 + hif * 16;
      union { v16h v; v8h h2[2]; } u;
      u.h2[0] = *(const v8h*)(p);
      u.h2[1] = *(const v8h*)(p + 8);
      afr[mb] = u.v;
    }
    // B fragments: lanes 0-15 hold K 0..15, lanes 16-31 hold K 16..31 (contiguous)
    for (int nb = 0; nb < 2; ++nb) {
      const _Float16* p = sB + (wn * 32 + nb * 16 + row) * 32 + hif * 16;
      bfr[nb] = *(const v16h*)(p);
    }
    for (int mb = 0; mb < 2; ++mb)
      for (int nb = 0; nb < 2; ++nb)
        acc[mb][nb] = __builtin_amdgcn_wmma_f32_16x16x32_f16(
            false, afr[mb], false, bfr[nb], (short)0, acc[mb][nb], false, false);
  }

  // C layout: VGPR j -> M = j + 8*(lane>=16), N = lane%16
  for (int mb = 0; mb < 2; ++mb)
    for (int nb = 0; nb < 2; ++nb)
      for (int j = 0; j < 8; ++j) {
        int m = m0 + wm * 32 + mb * 16 + j + hif * 8;
        int n = n0 + wn * 32 + nb * 16 + row;
        C[(size_t)m * N + n] = acc[mb][nb][j];
      }
}

// ---------------------------------------------------------------------------
// RoPE + pack: fp32 [B,S,H,Dk] -> f16 [B,H,S,Dk]; RoPE on Q,K; plain cvt on V.
// ---------------------------------------------------------------------------
__global__ __launch_bounds__(256)
void rope_pack(const float* __restrict__ qf, const float* __restrict__ kf,
               const float* __restrict__ vf, const int* __restrict__ pos,
               _Float16* __restrict__ qh, _Float16* __restrict__ kh,
               _Float16* __restrict__ vh) {
  int t = blockIdx.x * 256 + threadIdx.x;
  int i = t & 31;  t >>= 5;    // pair index 0..31
  int hd = t & 15; t >>= 4;    // head
  int s = t & 2047; t >>= 11;  // seq
  int b = t;                   // batch
  float p = (float)pos[s];
  // inv_freq = THETA^(-2i/Dk) = exp(-(2i/64) * ln(10000))
  float freq = __expf(-(2.0f * (float)i / 64.0f) * 9.210340371976184f);
  float sn, cs;
  __sincosf(p * freq, &sn, &cs);
  size_t src = ((size_t)(b * SEQ + s) * NUM_HEADS + hd) * DK + 2 * i;
  size_t dst = ((size_t)(b * NUM_HEADS + hd) * SEQ + s) * DK + 2 * i;
  float q1 = qf[src], q2 = qf[src + 1];
  qh[dst]     = (_Float16)(q1 * cs - q2 * sn);
  qh[dst + 1] = (_Float16)(q1 * sn + q2 * cs);
  float k1 = kf[src], k2 = kf[src + 1];
  kh[dst]     = (_Float16)(k1 * cs - k2 * sn);
  kh[dst + 1] = (_Float16)(k1 * sn + k2 * cs);
  vh[dst]     = (_Float16)vf[src];
  vh[dst + 1] = (_Float16)vf[src + 1];
}

// ---------------------------------------------------------------------------
// Async copy helper: global f16 -> LDS, 16 B per lane, tracked by ASYNCcnt.
// LDS operand is the low 32 bits of the generic shared-memory address
// ---------------------------------------------------------------------------
__device__ __forceinline__ void async_copy_b128(const _Float16* g, _Float16* l) {
  unsigned lds_off = (unsigned)(uintptr_t)l;
  unsigned long long ga = (unsigned long long)(uintptr_t)g;
  asm volatile("global_load_async_to_lds_b128 %0, %1, off"
               :: "v"(lds_off), "v"(ga) : "memory");
}
__device__ __forceinline__ void async_wait0() {
  asm volatile("s_wait_asynccnt 0" ::: "memory");
}

// ---------------------------------------------------------------------------
// Flash attention (causal, online softmax). Grid (S/64, B*H), 128 thr = 4 waves.
// Each wave owns 16 query rows; k processed 32 per iteration so the PV WMMAs
// run with a fully dense K=32 contraction (no zero padding).
// Scores computed transposed (S^T = K * Q^T): softmax stats are an in-lane
// 16-element reduce + one shfl_xor(16). Q and K tiles staged into LDS with
// GLOBAL_LOAD_ASYNC_TO_LDS_B128 (ASYNCcnt); V staged transposed by VALU.
// ---------------------------------------------------------------------------
__global__ __launch_bounds__(128)
void flash_attn(const _Float16* __restrict__ qh, const _Float16* __restrict__ kh,
                const _Float16* __restrict__ vh, float* __restrict__ O) {
  __shared__ _Float16 sQ[64 * 64];     // [q_local][d]
  __shared__ _Float16 sK[32 * 64];     // [k][d]
  __shared__ _Float16 sVT[64 * 32];    // [d][k]
  __shared__ _Float16 sP[4 * 16 * 32]; // per-wave P[q][k], k = 0..31
  const int tid  = threadIdx.x;
  const int lane = tid & 31;
  const int w    = tid >> 5;
  const int row  = lane & 15;
  const int hif  = lane >> 4;
  const int bh   = blockIdx.y;
  const int q0   = blockIdx.x * 64;
  const size_t base = (size_t)bh * SEQ * DK;

  // stage Q block: flat async copy of 4096 halfs (4 x b128 per lane)
  {
    const _Float16* src = qh + base + (size_t)q0 * DK;
    for (int c = tid; c < 512; c += 128)
      async_copy_b128(src + c * 8, sQ + c * 8);
  }
  async_wait0();
  __syncthreads();

  v8f acc[4] = {};
  float mrun = -1e30f, lrun = 0.0f;
  const int qw = q0 + w * 16;
  const int qg = qw + row;                  // this lane's global query row
  const int nt = (q0 >> 5) + 2;             // causal 32-k tile count (uniform)
  _Float16* myP = sP + w * 16 * 32;

  for (int kt = 0; kt < nt; ++kt) {
    __syncthreads();
    // async-stage K tile (32x64 halfs, flat copy, 2 x b128 per lane)
    {
      const _Float16* src = kh + base + (size_t)kt * 32 * DK;
      for (int c = tid; c < 256; c += 128)
        async_copy_b128(src + c * 8, sK + c * 8);
    }
    // stage V transposed: sVT[d][k] = V[k][d]  (overlaps with async K copy)
    {
      int k = tid & 31, dc = (tid >> 5) * 16;
      const _Float16* src = vh + base + (size_t)(kt * 32 + k) * DK + dc;
      v8h v0 = *(const v8h*)src;
      v8h v1 = *(const v8h*)(src + 8);
      for (int j = 0; j < 8; ++j) sVT[(dc + j) * 32 + k] = v0[j];
      for (int j = 0; j < 8; ++j) sVT[(dc + 8 + j) * 32 + k] = v1[j];
    }
    async_wait0();
    __syncthreads();

    // S^T = K * Q^T : A = K (m = k-row, contraction over d), B = Q^T (n = q)
    const _Float16* pq = sQ + (w * 16 + row) * 64 + hif * 16;
    v16h bQ0 = *(const v16h*)(pq);
    v16h bQ1 = *(const v16h*)(pq + 32);
    v8f sfrag[2];
    for (int kb = 0; kb < 2; ++kb) {
      const _Float16* pk = sK + (kb * 16 + row) * 64 + hif * 16;
      union { v16h v; v8h h2[2]; } u;
      u.h2[0] = *(const v8h*)(pk);      u.h2[1] = *(const v8h*)(pk + 8);
      v16h aK0 = u.v;
      u.h2[0] = *(const v8h*)(pk + 32); u.h2[1] = *(const v8h*)(pk + 40);
      v16h aK1 = u.v;
      v8f s = {};
      s = __builtin_amdgcn_wmma_f32_16x16x32_f16(false, aK0, false, bQ0, (short)0, s, false, false);
      s = __builtin_amdgcn_wmma_f32_16x16x32_f16(false, aK1, false, bQ1, (short)0, s, false, false);
      sfrag[kb] = s;
    }

    // online softmax: element (kb, j) holds k = kt*32 + kb*16 + j + 8*hif, q = qg
    float sc[2][8];
    float mt = -1e30f;
    for (int kb = 0; kb < 2; ++kb)
      for (int j = 0; j < 8; ++j) {
        int kgl = kt * 32 + kb * 16 + j + hif * 8;
        float v = sfrag[kb][j] * 0.125f;       // 1/sqrt(64)
        if (kgl > qg) v = -1e30f;              // causal mask
        sc[kb][j] = v;
        mt = fmaxf(mt, v);
      }
    mt = fmaxf(mt, __shfl_xor(mt, 16, 32));    // lane pair (L, L+16) share q
    float mnew  = fmaxf(mrun, mt);
    float alpha = __expf(mrun - mnew);
    float psum = 0.0f;
    for (int kb = 0; kb < 2; ++kb)
      for (int j = 0; j < 8; ++j) {
        float p = __expf(sc[kb][j] - mnew);
        psum += p;
        myP[row * 32 + kb * 16 + j + hif * 8] = (_Float16)p;  // P[q][k] row-major
      }
    psum += __shfl_xor(psum, 16, 32);
    lrun = lrun * alpha + psum;
    mrun = mnew;
    for (int t4 = 0; t4 < 4; ++t4)
      for (int j = 0; j < 8; ++j) acc[t4][j] *= alpha;

    // keep the cross-lane LDS stores ordered before the loads (LDS in-order per wave)
    __asm__ volatile("" ::: "memory");

    // O^T += V^T * P^T : A = V^T (m = d), B = P^T (n = q), dense K=32
    v16h bP = *(const v16h*)(myP + row * 32 + hif * 16);
    for (int db = 0; db < 4; ++db) {
      const _Float16* pv = sVT + (db * 16 + row) * 32 + hif * 16;
      union { v16h v; v8h h2[2]; } u;
      u.h2[0] = *(const v8h*)(pv);
      u.h2[1] = *(const v8h*)(pv + 8);
      acc[db] = __builtin_amdgcn_wmma_f32_16x16x32_f16(
          false, u.v, false, bP, (short)0, acc[db], false, false);
    }
  }

  // write O (fragment: d = db*16 + j + 8*hif, q = qg), layout [B,S,H*Dk] fp32
  const int b  = bh >> 4;
  const int hd = bh & 15;
  float inv_l = 1.0f / lrun;
  for (int db = 0; db < 4; ++db)
    for (int j = 0; j < 8; ++j) {
      int d = db * 16 + j + hif * 8;
      O[((size_t)(b * SEQ) + qg) * D_MODEL + hd * DK + d] = acc[db][j] * inv_l;
    }
}

// ---------------------------------------------------------------------------
extern "C" void kernel_launch(void* const* d_in, const int* in_sizes, int n_in,
                              void* d_out, int out_size, void* d_ws, size_t ws_size,
                              hipStream_t stream) {
  (void)in_sizes; (void)n_in; (void)out_size; (void)ws_size;
  const float* x   = (const float*)d_in[0];
  const int*   pos = (const int*)d_in[1];
  const float* Wq  = (const float*)d_in[2];
  const float* Wk  = (const float*)d_in[3];
  const float* Wv  = (const float*)d_in[4];
  const float* Wo  = (const float*)d_in[5];
  float* out = (float*)d_out;

  char* ws = (char*)d_ws;
  const size_t SZF = (size_t)MTOT * D_MODEL * sizeof(float);    // 16 MB
  const size_t SZH = (size_t)MTOT * D_MODEL * sizeof(_Float16); //  8 MB
  float*    qf = (float*)(ws);
  float*    kf = (float*)(ws + SZF);
  float*    vf = (float*)(ws + 2 * SZF);
  _Float16* qh = (_Float16*)(ws + 3 * SZF);
  _Float16* kh = (_Float16*)(ws + 3 * SZF + SZH);
  _Float16* vh = (_Float16*)(ws + 3 * SZF + 2 * SZH);
  float*    of = qf;  // reuse Q fp32 buffer for attention output

  dim3 gg(D_MODEL / 128, MTOT / 64);
  wmma_gemm_nt<<<gg, 256, 0, stream>>>(x, Wq, qf, MTOT, D_MODEL, D_MODEL);
  wmma_gemm_nt<<<gg, 256, 0, stream>>>(x, Wk, kf, MTOT, D_MODEL, D_MODEL);
  wmma_gemm_nt<<<gg, 256, 0, stream>>>(x, Wv, vf, MTOT, D_MODEL, D_MODEL);
  rope_pack<<<dim3((BATCH * SEQ * NUM_HEADS * 32) / 256), 256, 0, stream>>>(
      qf, kf, vf, pos, qh, kh, vh);
  flash_attn<<<dim3(SEQ / 64, BATCH * NUM_HEADS), 128, 0, stream>>>(qh, kh, vh, of);
  wmma_gemm_nt<<<gg, 256, 0, stream>>>(of, Wo, out, MTOT, D_MODEL, D_MODEL);
}
